// MultiQueryAttention_18657337933978
// MI455X (gfx1250) — matrix operand order repaired and checked by reading, not compiled, not run on hardware
//
#include <hip/hip_runtime.h>
#include <hip/hip_bf16.h>

#define B_      2
#define TQ_     2048
#define TKV_    2048
#define DMODEL_ 1024
#define NH_     16
#define DH_     64

typedef __attribute__((ext_vector_type(16))) __bf16 v16bf;
typedef __attribute__((ext_vector_type(8)))  __bf16 v8bf;
typedef __attribute__((ext_vector_type(8)))  float  v8f;

// ---- WMMA fragment helpers (CDNA5 wave32 layouts, CDNA5 ISA §7.12.2) ----

__device__ __forceinline__ v8f wmma_bf16(v16bf a, v16bf b, v8f c) {
    return __builtin_amdgcn_wmma_f32_16x16x32_bf16(false, a, false, b, (short)0, c, false, false);
}

// A-matrix 16x32 bf16: lane L holds row M=L&15; half=L>>4 selects K chunks
// {k0+0..7, k0+16..23} (half=0) or {k0+8..15, k0+24..31} (half=1).
__device__ __forceinline__ v16bf load_a_bf16(const __bf16* A, int lda, int m0, int k0, int lane) {
    int m    = m0 + (lane & 15);
    int half = lane >> 4;
    const __bf16* p = A + (size_t)m * lda + k0 + half * 8;
    v8bf lo = *(const v8bf*)(p);
    v8bf hi = *(const v8bf*)(p + 16);
    v16bf a;
#pragma unroll
    for (int i = 0; i < 8; ++i) { a[i] = lo[i]; a[i + 8] = hi[i]; }
    return a;
}

// B-matrix 32x16 bf16 (K x N): lane L holds column N=n0+(L&15),
// K chunks {k0+0..7, k0+16..23} (half=0) or {k0+8..15, k0+24..31} (half=1).  W is row-major [N, K].
__device__ __forceinline__ v16bf load_b_bf16(const __bf16* W, int ldw, int n0, int k0, int lane) {
    int n = n0 + (lane & 15);
    const __bf16* p = W + (size_t)n * ldw + k0 + (lane >> 4) * 8;
    v8bf lo = *(const v8bf*)(p);
    v8bf hi = *(const v8bf*)(p + 16);
    v16bf b;
#pragma unroll
    for (int i = 0; i < 8; ++i) { b[i] = lo[i]; b[i + 8] = hi[i]; }
    return b;
}

// ---- elementwise converts ----

__global__ __launch_bounds__(256) void f32_to_bf16_kernel(const float* __restrict__ in,
                                                          __bf16* __restrict__ out, int n) {
    int i = blockIdx.x * 256 + threadIdx.x;
    if (i < n) out[i] = (__bf16)in[i];
}

// Vbuf [B*TKV, DH] bf16 -> Vt [B, DH, TKV] bf16
__global__ __launch_bounds__(256) void transpose_v_kernel(const __bf16* __restrict__ Vb,
                                                          __bf16* __restrict__ Vt) {
    int idx = blockIdx.x * 256 + threadIdx.x;        // over B*TKV*DH
    int d = idx & (DH_ - 1);
    int t = (idx >> 6) & (TKV_ - 1);
    int b = idx >> 17;                               // 6 + 11 bits
    Vt[((size_t)b * DH_ + d) * TKV_ + t] = Vb[idx];
}

// ---- generic GEMM: out[M,N] = A[M,K] @ W[N,K]^T + bias ----
// All call sites have lda == K, so the block's 16-row A strip is one contiguous
// 16*K*2 = 32 KB region: async-copy it to LDS once (ASYNCcnt path), then every
// wave builds A-fragments from LDS instead of 4x-redundant global loads.
// grid.x = N/64 (4 waves x 16 cols), grid.y = M/16, block = 128 (4 waves)

template <bool OUT_F32>
__global__ __launch_bounds__(128) void gemm_bias_kernel(const __bf16* __restrict__ A,
                                                        const __bf16* __restrict__ W,
                                                        const float* __restrict__ bias,
                                                        void* __restrict__ outp,
                                                        int K, int lda, int ldw, int ldo) {
    __shared__ __bf16 As[16 * DMODEL_];     // 32 KB A strip (K <= DMODEL_)

    int lane = threadIdx.x & 31;
    int wave = threadIdx.x >> 5;
    int n0 = blockIdx.x * 64 + wave * 16;
    int m0 = blockIdx.y * 16;

    // ---- async stage A strip: global -> LDS (global_load_async_to_lds_b128) ----
    {
        const char* srcA = (const char*)(A + (size_t)m0 * lda);   // contiguous (lda==K)
        unsigned ldsbase = (unsigned)(uintptr_t)(&As[0]);
        int bytes = 16 * K * 2;
        for (int off = threadIdx.x * 16; off < bytes; off += 128 * 16) {
            unsigned laddr = ldsbase + off;
            unsigned long long gaddr = (unsigned long long)(srcA + off);
            asm volatile("global_load_async_to_lds_b128 %0, %1, off"
                         :: "v"(laddr), "v"(gaddr) : "memory");
        }
        asm volatile("s_wait_asynccnt 0x0" ::: "memory");
        __syncthreads();
    }

    const __bf16* Al = (const __bf16*)As;   // strip-local, row stride K
    v8f acc = {0.f, 0.f, 0.f, 0.f, 0.f, 0.f, 0.f, 0.f};
    for (int k0 = 0; k0 < K; k0 += 32) {
        v16bf a = load_a_bf16(Al, K, 0, k0, lane);   // ds_load_b128 from LDS
        v16bf b = load_b_bf16(W, ldw, n0, k0, lane);
        acc = wmma_bf16(a, b, acc);
    }

    int n    = n0 + (lane & 15);
    int half = lane >> 4;
    float bv = bias[n];
#pragma unroll
    for (int v = 0; v < 8; ++v) {
        int m = m0 + v + 8 * half;
        float val = acc[v] + bv;
        if (OUT_F32) ((float*)outp)[(size_t)m * ldo + n] = val;
        else         ((__bf16*)outp)[(size_t)m * ldo + n] = (__bf16)val;
    }
}

// ---- fused attention: per (b,h,16-row q strip): scores + mask + softmax + P.V ----
// One 16 x TKV fp32 score strip lives in LDS (128 KB).  attn_weights written to
// HBM exactly once.  grid = (TQ/16, B*NH), block = 128 (4 waves).

__global__ __launch_bounds__(128) void attn_fused_kernel(const __bf16* __restrict__ Qb,
                                                         const __bf16* __restrict__ Kb,
                                                         const __bf16* __restrict__ Vt,
                                                         float* __restrict__ attn,
                                                         __bf16* __restrict__ attended) {
    extern __shared__ float smem[];
    float* Ps       = smem;                 // [16][TKV] score/prob strip
    float* part_red = smem + 16 * TKV_;     // [16][8] partial reductions
    float* row_stat = part_red + 16 * 8;    // [16] row max, then row sum

    int lane = threadIdx.x & 31;
    int wave = threadIdx.x >> 5;
    int half = lane >> 4;
    int nlo  = lane & 15;
    int m0 = blockIdx.x * 16;               // q strip base
    int bh = blockIdx.y;
    int b = bh / NH_, h = bh % NH_;

    const __bf16* Aq = Qb + (size_t)b * TQ_ * DMODEL_ + h * DH_;  // lda = DMODEL
    const __bf16* Kw = Kb + (size_t)b * TKV_ * DH_;               // ldw = DH
    const __bf16* Vw = Vt + (size_t)b * DH_ * TKV_;               // ldw = TKV

    // Q strip A-fragments (16x64 bf16) pinned in VGPRs for the whole key loop
    v16bf qa0 = load_a_bf16(Aq, DMODEL_, m0, 0,  lane);
    v16bf qa1 = load_a_bf16(Aq, DMODEL_, m0, 32, lane);

    // ---- phase 1: scores -> LDS.  wave w covers key tiles w*16, 64+w*16, ... ----
    const float scale = 0.125f;             // 1/sqrt(64)
    for (int n0 = wave * 16; n0 < TKV_; n0 += 64) {
        if (n0 + 64 < TKV_)
            __builtin_prefetch(Kw + (size_t)(n0 + 64) * DH_, 0, 1);  // global_prefetch
        v8f acc = {0.f, 0.f, 0.f, 0.f, 0.f, 0.f, 0.f, 0.f};
        acc = wmma_bf16(qa0, load_b_bf16(Kw, DH_, n0, 0,  lane), acc);
        acc = wmma_bf16(qa1, load_b_bf16(Kw, DH_, n0, 32, lane), acc);
        int n = n0 + nlo;
#pragma unroll
        for (int v = 0; v < 8; ++v) {
            int m = v + 8 * half;           // row within strip
            float val = acc[v] * scale;
            if (n > m0 + m) val = -__builtin_inff();   // causal mask (triu, k=1)
            Ps[m * TKV_ + n] = val;
        }
    }
    __syncthreads();

    // ---- phase 2: row softmax in LDS (8 threads per row, 64 float4 each) ----
    int row  = threadIdx.x >> 3;
    int part = threadIdx.x & 7;
    float4* pr4 = (float4*)(Ps + row * TKV_);

    float mx = -__builtin_inff();
    for (int i = part * 64; i < part * 64 + 64; ++i) {
        float4 v = pr4[i];
        mx = fmaxf(fmaxf(fmaxf(mx, v.x), v.y), fmaxf(v.z, v.w));
    }
    part_red[row * 8 + part] = mx;
    __syncthreads();
    if (part == 0) {
        float m2 = part_red[row * 8];
#pragma unroll
        for (int i = 1; i < 8; ++i) m2 = fmaxf(m2, part_red[row * 8 + i]);
        row_stat[row] = m2;
    }
    __syncthreads();
    mx = row_stat[row];

    float sum = 0.f;
    for (int i = part * 64; i < part * 64 + 64; ++i) {
        float4 v = pr4[i];
        v.x = __expf(v.x - mx); v.y = __expf(v.y - mx);
        v.z = __expf(v.z - mx); v.w = __expf(v.w - mx);
        pr4[i] = v;
        sum += (v.x + v.y) + (v.z + v.w);
    }
    part_red[row * 8 + part] = sum;
    __syncthreads();
    if (part == 0) {
        float s2 = 0.f;
#pragma unroll
        for (int i = 0; i < 8; ++i) s2 += part_red[row * 8 + i];
        row_stat[row] = s2;                 // now holds row sum
    }
    __syncthreads();

    // ---- phase 3: normalize + single coalesced write of attn_weights ----
    float4* Ps4  = (float4*)Ps;
    float4* out4 = (float4*)(attn + ((size_t)bh * TQ_ + m0) * TKV_);
    for (int j = threadIdx.x; j < 16 * TKV_ / 4; j += 128) {
        int r = j >> 9;                     // 512 float4 per row
        float inv = 1.0f / row_stat[r];
        float4 v = Ps4[j];
        v.x *= inv; v.y *= inv; v.z *= inv; v.w *= inv;
        Ps4[j] = v;
        out4[j] = v;
    }
    __syncthreads();

    // ---- phase 4: attended strip = P @ V from LDS.  wave w covers d-tile w*16 ----
    int n0 = wave * 16;
    v8f acc = {0.f, 0.f, 0.f, 0.f, 0.f, 0.f, 0.f, 0.f};
    for (int k0 = 0; k0 < TKV_; k0 += 32) {
        const float* p = Ps + (size_t)nlo * TKV_ + k0 + half * 8;
        v16bf a;
#pragma unroll
        for (int i = 0; i < 8; ++i) { a[i] = (__bf16)p[i]; a[i + 8] = (__bf16)p[i + 16]; }
        v16bf bb = load_b_bf16(Vw, TKV_, n0, k0, lane);
        acc = wmma_bf16(a, bb, acc);
    }
    int n = n0 + nlo;
#pragma unroll
    for (int v = 0; v < 8; ++v) {
        int m = m0 + v + 8 * half;
        attended[((size_t)(b * TQ_ + m)) * DMODEL_ + h * DH_ + n] = (__bf16)acc[v];
    }
}

// ---------------------------------------------------------------------------

extern "C" void kernel_launch(void* const* d_in, const int* in_sizes, int n_in,
                              void* d_out, int out_size, void* d_ws, size_t ws_size,
                              hipStream_t stream) {
    const float* q  = (const float*)d_in[0];
    const float* kv = (const float*)d_in[1];
    // d_in[2] = mask: fixed causal triu, computed analytically in attn_fused_kernel
    const float* Wq = (const float*)d_in[3];
    const float* bq = (const float*)d_in[4];
    const float* Wk = (const float*)d_in[5];
    const float* bk = (const float*)d_in[6];
    const float* Wv = (const float*)d_in[7];
    const float* bv = (const float*)d_in[8];
    const float* Wo = (const float*)d_in[9];
    const float* bo = (const float*)d_in[10];

    float* out  = (float*)d_out;                               // [B*TQ, DMODEL]
    float* attn = out + (size_t)B_ * TQ_ * DMODEL_;            // [B*NH, TQ, TKV]

    const int nQ  = B_ * TQ_ * DMODEL_;     // 4,194,304 (q, kv, Qbuf, attended)
    const int nWq = DMODEL_ * DMODEL_;      // 1,048,576 (Wq, Wo)
    const int nWk = DH_ * DMODEL_;          // 65,536    (Wk, Wv)
    const int nKV = B_ * TKV_ * DH_;        // 262,144   (Kbuf, Vbuf, Vt)

    char* ws = (char*)d_ws;
    size_t off = 0;
    auto alloc = [&](size_t elems) -> __bf16* {
        __bf16* p = (__bf16*)(ws + off);
        off = (off + elems * sizeof(__bf16) + 255) & ~(size_t)255;
        return p;
    };
    __bf16* q_bf   = alloc(nQ);
    __bf16* kv_bf  = alloc(nQ);
    __bf16* Wq_bf  = alloc(nWq);
    __bf16* Wk_bf  = alloc(nWk);
    __bf16* Wv_bf  = alloc(nWk);
    __bf16* Wo_bf  = alloc(nWq);
    __bf16* Qbuf   = alloc(nQ);
    __bf16* Kbuf   = alloc(nKV);
    __bf16* Vbuf   = alloc(nKV);
    __bf16* Vt     = alloc(nKV);
    __bf16* attnd  = alloc(nQ);

    // 1) fp32 -> bf16 converts
    f32_to_bf16_kernel<<<nQ  / 256, 256, 0, stream>>>(q,  q_bf,  nQ);
    f32_to_bf16_kernel<<<nQ  / 256, 256, 0, stream>>>(kv, kv_bf, nQ);
    f32_to_bf16_kernel<<<nWq / 256, 256, 0, stream>>>(Wq, Wq_bf, nWq);
    f32_to_bf16_kernel<<<nWk / 256, 256, 0, stream>>>(Wk, Wk_bf, nWk);
    f32_to_bf16_kernel<<<nWk / 256, 256, 0, stream>>>(Wv, Wv_bf, nWk);
    f32_to_bf16_kernel<<<nWq / 256, 256, 0, stream>>>(Wo, Wo_bf, nWq);

    const int MR = B_ * TQ_;  // 4096 rows for every row-major GEMM

    // 2) projections: Q = q@Wq^T+bq; K = kv@Wk^T+bk; V = kv@Wv^T+bv
    gemm_bias_kernel<false><<<dim3(DMODEL_ / 64, MR / 16), 128, 0, stream>>>(
        q_bf, Wq_bf, bq, Qbuf, DMODEL_, DMODEL_, DMODEL_, DMODEL_);
    gemm_bias_kernel<false><<<dim3(DH_ / 64, MR / 16), 128, 0, stream>>>(
        kv_bf, Wk_bf, bk, Kbuf, DMODEL_, DMODEL_, DMODEL_, DH_);
    gemm_bias_kernel<false><<<dim3(DH_ / 64, MR / 16), 128, 0, stream>>>(
        kv_bf, Wv_bf, bv, Vbuf, DMODEL_, DMODEL_, DMODEL_, DH_);

    // 3) V -> V^T per batch for contiguous-K WMMA B fragments in P.V
    transpose_v_kernel<<<nKV / 256, 256, 0, stream>>>(Vbuf, Vt);

    // 4) fused scores + mask + softmax + P.V (single write of attn_weights)
    size_t smem = (size_t)(16 * TKV_ + 16 * 8 + 16) * sizeof(float);   // ~128.6 KB
    (void)hipFuncSetAttribute((const void*)attn_fused_kernel,
                              hipFuncAttributeMaxDynamicSharedMemorySize, (int)smem);
    attn_fused_kernel<<<dim3(TQ_ / 16, B_ * NH_), 128, smem, stream>>>(
        Qbuf, Kbuf, Vt, attn, attnd);

    // 5) out = attended @ Wo^T + bo  (fp32 output)
    gemm_bias_kernel<true><<<dim3(DMODEL_ / 64, MR / 16), 128, 0, stream>>>(
        attnd, Wo_bf, bo, out, DMODEL_, DMODEL_, DMODEL_, DMODEL_);
}
